// ModulatedTransposedConvolutionLayer_88192858456285
// MI455X (gfx1250) — compile-verified, matrix operand order checked by reading, not run on hardware
//
#include <hip/hip_runtime.h>
#include <hip/hip_bf16.h>
#include <math.h>

// ---------------------------------------------------------------------------
// ModulatedTransposedConvolutionLayer for MI455X (gfx1250, wave32, WMMA)
//
// y = blur5x5( coef[b,o] * implicit_gemm(style[b,i]*x_norm, weight) + bias ) -> lrelu*sqrt(2)
//
// Main conv = 4 parity-class implicit GEMMs (M=16384, N=256, K=512*taps) using
// v_wmma_f32_16x16x32_bf16 with double-buffered LDS staging (1 barrier/chunk).
// ---------------------------------------------------------------------------

typedef __attribute__((ext_vector_type(16))) __bf16 v16bf;
typedef __attribute__((ext_vector_type(8)))  float  v8f;

#define B_    16
#define CIN_  512
#define COUT_ 256
#define HW_   32
#define EPS_  1e-8f

__device__ __forceinline__ unsigned short f2bf(float f) {
  // round-to-nearest-even fp32 -> bf16
  unsigned int u = __float_as_uint(f);
  unsigned int r = u + 0x7FFFu + ((u >> 16) & 1u);
  return (unsigned short)(r >> 16);
}

struct Frag32 { uint4 lo; uint4 hi; };   // 32 bytes == 16 bf16 == one WMMA A/B fragment per lane

// ---------------------------------------------------------------------------
// K1: style[b,i] = (s @ mod_w^T)/sqrt(512) + mod_b
// ---------------------------------------------------------------------------
__global__ void style_kernel(const float* __restrict__ s, const float* __restrict__ mod_w,
                             const float* __restrict__ mod_b, float* __restrict__ style) {
  int idx = blockIdx.x * blockDim.x + threadIdx.x;
  if (idx >= B_ * CIN_) return;
  int b = idx >> 9, i = idx & 511;
  const float* sr = s + (size_t)b * CIN_;
  const float* wr = mod_w + (size_t)i * CIN_;
  float acc = 0.f;
  for (int j = 0; j < CIN_; ++j) acc += sr[j] * wr[j];
  style[idx] = acc * 0.04419417382415922f /* 1/sqrt(512) */ + mod_b[i];
}

// ---------------------------------------------------------------------------
// K2: repack weight [CIN,COUT,3,3] -> Wt[kh*3+kw][o][i] (bf16, i contiguous)
//     and Wsq[o][i] = sum_{kh,kw} weight^2
// ---------------------------------------------------------------------------
__global__ void prep_w_kernel(const float* __restrict__ weight,
                              unsigned short* __restrict__ Wt, float* __restrict__ Wsq) {
  int idx = blockIdx.x * blockDim.x + threadIdx.x;
  if (idx >= CIN_ * COUT_) return;
  int o = idx & 255, i = idx >> 8;
  const float* wr = weight + ((size_t)i * COUT_ + o) * 9;
  float ss = 0.f;
#pragma unroll
  for (int t = 0; t < 9; ++t) {
    float v = wr[t];
    ss += v * v;
    Wt[((size_t)t * COUT_ + o) * CIN_ + i] = f2bf(v);
  }
  Wsq[(size_t)o * CIN_ + i] = ss;
}

// ---------------------------------------------------------------------------
// K3: xs[b][h][w][i] = bf16( x_norm * style[b,i] ), padded to 33x33 with zeros
//     x_norm = x * rsqrt(mean_i x^2 + eps)       (one block per (h,w,b) pixel)
// ---------------------------------------------------------------------------
__global__ void xs_kernel(const float* __restrict__ x, const float* __restrict__ style,
                          unsigned short* __restrict__ xs) {
  int h = blockIdx.x, w = blockIdx.y, b = blockIdx.z;
  int tid = threadIdx.x;
  size_t dst = (((size_t)b * 33 + h) * 33 + w) * CIN_;
  if (h == 32 || w == 32) {            // zero padding ring (uniform branch per block)
    xs[dst + tid] = 0; xs[dst + tid + 256] = 0;
    return;
  }
  float x0 = x[(((size_t)b * CIN_ + tid)       * HW_ + h) * HW_ + w];
  float x1 = x[(((size_t)b * CIN_ + tid + 256) * HW_ + h) * HW_ + w];
  __shared__ float red[256];
  red[tid] = x0 * x0 + x1 * x1;
  __syncthreads();
  for (int st = 128; st > 0; st >>= 1) {
    if (tid < st) red[tid] += red[tid + st];
    __syncthreads();
  }
  float scale = rsqrtf(red[0] * (1.f / CIN_) + EPS_);
  xs[dst + tid]       = f2bf(x0 * scale * style[(size_t)b * CIN_ + tid]);
  xs[dst + tid + 256] = f2bf(x1 * scale * style[(size_t)b * CIN_ + tid + 256]);
}

// ---------------------------------------------------------------------------
// K4: coef[b,o] = rsqrt( (sum_i style^2 * Wsq) / 4608 + eps ) / sqrt(4608)
// ---------------------------------------------------------------------------
__global__ void coef_kernel(const float* __restrict__ style, const float* __restrict__ Wsq,
                            float* __restrict__ coef) {
  int b = blockIdx.x, o = threadIdx.x;
  const float* st = style + (size_t)b * CIN_;
  const float* wq = Wsq + (size_t)o * CIN_;
  float s = 0.f;
  for (int i = 0; i < CIN_; ++i) { float sv = st[i]; s += sv * sv * wq[i]; }
  float demod = rsqrtf(s * (1.f / 4608.f) + EPS_);
  coef[(size_t)b * COUT_ + o] = demod * 0.014731391274719739f; // 1/sqrt(4608)
}

// ---------------------------------------------------------------------------
// K5: main implicit GEMM per parity class (blockIdx.z = class, pp = z&1, pq = z>>1)
//     y[p,q] (p=2ph+pp, q=2pw+pq) = coef*sum_{taps,i} xs[b,ph+dh,pw+dw,i]*Wt[kh,kw,o,i] + bias
//     M tile 64, N tile 256 (full), K chunks of 32, 8 waves (2 M x 4 N), 8 wmma per chunk,
//     double-buffered LDS: next chunk's global loads issued before current chunk's WMMAs.
// ---------------------------------------------------------------------------
__global__ __launch_bounds__(256) void gemm_kernel(const unsigned short* __restrict__ xs,
                                                   const unsigned short* __restrict__ Wt,
                                                   const float* __restrict__ coef,
                                                   const float* __restrict__ bias,
                                                   float* __restrict__ ypre) {
  __shared__ unsigned short At[2][64][32];    //  2 x 4 KB
  __shared__ unsigned short Bt[2][256][32];   //  2 x 16 KB

  const int tid  = threadIdx.x;
  const int lane = tid & 31, wid = tid >> 5;
  const int wm = wid & 1;        // wave position in M (2)
  const int wn = wid >> 1;       // wave position in N (4)
  const int cls = blockIdx.z;
  const int pp = cls & 1, pq = cls >> 1;
  const int m0 = blockIdx.x * 64;

  // A-load assignment: 256 threads -> 64 rows x 4 segments of 16B
  const int arow = tid >> 2, aseg = tid & 3;
  const int am  = m0 + arow;
  const int aph = (am >> 5) & 31, apw = am & 31;
  const int ab  = am >> 10;

  // parity-class tap tables: p = 2h - 1 + kh  =>  even p: kh=1 (h=ph); odd p: kh=0 (h=ph+1), kh=2 (h=ph)
  int khs[2] = {1, 1}, dhs[2] = {0, 0}, kws[2] = {1, 1}, dws[2] = {0, 0};
  int nth = 1, ntw = 1;
  if (pp) { nth = 2; khs[0] = 0; dhs[0] = 1; khs[1] = 2; dhs[1] = 0; }
  if (pq) { ntw = 2; kws[0] = 0; dws[0] = 1; kws[1] = 2; dws[1] = 0; }
  const int ntap = nth * ntw;

  // per-tap base pointers (flattened tap index t = th*ntw + tw)
  const unsigned short* aBaseTap[4];
  const unsigned short* bBaseTap[4];
  for (int t = 0; t < 4; ++t) {
    const int tt = (t < ntap) ? t : 0;
    const int th = tt / ntw, tw = tt % ntw;
    const int h = aph + dhs[th], w = apw + dws[tw];
    aBaseTap[t] = xs + (((size_t)ab * 33 + h) * 33 + w) * CIN_ + aseg * 8;
    bBaseTap[t] = Wt + ((size_t)(khs[th] * 3 + kws[tw]) * COUT_ + tid) * CIN_;
  }

  const int khalf = lane >> 4;
  const int lmod  = lane & 15;
  const int nchunk = ntap * 16;   // 16 chunks of 32 channels per tap

  // ---- software pipeline: global load (regs) for chunk c -------------------
  auto loadG = [&](int c, uint4& areg, uint4 (&breg)[4]) {
    const int tap = c >> 4, i0 = (c & 15) << 5;
    areg = *reinterpret_cast<const uint4*>(aBaseTap[tap] + i0);
    const uint4* bs = reinterpret_cast<const uint4*>(bBaseTap[tap] + i0);
    breg[0] = bs[0]; breg[1] = bs[1]; breg[2] = bs[2]; breg[3] = bs[3];
  };
  auto storeLds = [&](int buf, const uint4& areg, const uint4 (&breg)[4]) {
    *reinterpret_cast<uint4*>(&At[buf][arow][aseg * 8]) = areg;
    uint4* bd = reinterpret_cast<uint4*>(&Bt[buf][tid][0]);
    bd[0] = breg[0]; bd[1] = breg[1]; bd[2] = breg[2]; bd[3] = breg[3];
  };

  v8f acc[2][4] = {};
  uint4 aNext; uint4 bNext[4];
  loadG(0, aNext, bNext);
  storeLds(0, aNext, bNext);

  for (int c = 0; c < nchunk; ++c) {
    __syncthreads();                    // buf (c&1) ready; buf (c+1)&1 free to overwrite
    const int cur = c & 1;
    if (c + 1 < nchunk) {               // issue next chunk's loads ahead of this chunk's WMMAs
      loadG(c + 1, aNext, bNext);
      storeLds(cur ^ 1, aNext, bNext);
    }

    // fragments: lane m/n = lane&15, K split selected by lane>>4 (per ISA 16-bit A layout)
    v16bf afr[2], bfr[4];
#pragma unroll
    for (int mi = 0; mi < 2; ++mi) {
      const int m = wm * 32 + mi * 16 + lmod;
      Frag32 t;
      t.lo = *reinterpret_cast<const uint4*>(&At[cur][m][8 * khalf]);
      t.hi = *reinterpret_cast<const uint4*>(&At[cur][m][16 + 8 * khalf]);
      afr[mi] = __builtin_bit_cast(v16bf, t);
    }
#pragma unroll
    for (int ni = 0; ni < 4; ++ni) {
      const int n = wn * 64 + ni * 16 + lmod;
      Frag32 t;
      t.lo = *reinterpret_cast<const uint4*>(&Bt[cur][n][8 * khalf]);
      t.hi = *reinterpret_cast<const uint4*>(&Bt[cur][n][16 + 8 * khalf]);
      bfr[ni] = __builtin_bit_cast(v16bf, t);
    }
#pragma unroll
    for (int mi = 0; mi < 2; ++mi)
#pragma unroll
      for (int ni = 0; ni < 4; ++ni)
        acc[mi][ni] = __builtin_amdgcn_wmma_f32_16x16x32_bf16(
            false, afr[mi], false, bfr[ni], (short)0, acc[mi][ni], false, false);
  }

  // epilogue: D layout lane l: n = l&15, m = 8*(l>>4) + r  (r = vgpr index)
  const int b = m0 >> 10;
#pragma unroll
  for (int mi = 0; mi < 2; ++mi) {
#pragma unroll
    for (int ni = 0; ni < 4; ++ni) {
      const int n = wn * 64 + ni * 16 + lmod;
      const float cf = coef[(size_t)b * COUT_ + n];
      const float bs = bias[n];
#pragma unroll
      for (int r = 0; r < 8; ++r) {
        const int mt = wm * 32 + mi * 16 + 8 * khalf + r;
        const int m  = m0 + mt;
        const int ph = (m >> 5) & 31, pw = m & 31;
        const int p = 2 * ph + pp, q = 2 * pw + pq;
        ypre[(((size_t)b * COUT_ + n) * 64 + p) * 64 + q] = cf * acc[mi][ni][r] + bs;
      }
    }
  }
}

// ---------------------------------------------------------------------------
// K6: fused 5x5 binomial blur (zero pad) + leaky ReLU(0.2) * sqrt(2)
//     block = 16x16 output tile of one (b,o) plane, 20x20 LDS halo tile
// ---------------------------------------------------------------------------
__global__ __launch_bounds__(256) void blur_kernel(const float* __restrict__ ypre,
                                                   float* __restrict__ out) {
  __shared__ float tile[20][20];
  const int plane = blockIdx.z;                  // b*COUT + o
  const int tx = threadIdx.x, ty = threadIdx.y;
  const int ox0 = blockIdx.x * 16, oy0 = blockIdx.y * 16;
  const float* src = ypre + (size_t)plane * 64 * 64;

  const int lt = ty * 16 + tx;
  for (int k = lt; k < 400; k += 256) {
    int ly = k / 20, lx = k % 20;
    int gy = oy0 + ly - 2, gx = ox0 + lx - 2;
    tile[ly][lx] = (gy >= 0 && gy < 64 && gx >= 0 && gx < 64) ? src[gy * 64 + gx] : 0.f;
  }
  __syncthreads();

  const float k5[5] = {1.f / 16, 4.f / 16, 6.f / 16, 4.f / 16, 1.f / 16};
  float acc = 0.f;
#pragma unroll
  for (int dy = 0; dy < 5; ++dy) {
    float rs = 0.f;
#pragma unroll
    for (int dx = 0; dx < 5; ++dx) rs += k5[dx] * tile[ty + dy][tx + dx];
    acc += k5[dy] * rs;
  }
  float v = (acc >= 0.f) ? acc : 0.2f * acc;
  out[(size_t)plane * 4096 + (size_t)(oy0 + ty) * 64 + (ox0 + tx)] = v * 1.4142135623730951f;
}

// ---------------------------------------------------------------------------
// launch
// ---------------------------------------------------------------------------
extern "C" void kernel_launch(void* const* d_in, const int* in_sizes, int n_in,
                              void* d_out, int out_size, void* d_ws, size_t ws_size,
                              hipStream_t stream) {
  const float* x      = (const float*)d_in[0];   // [16,512,32,32]
  const float* s      = (const float*)d_in[1];   // [16,512]
  const float* weight = (const float*)d_in[2];   // [512,256,3,3]
  const float* bias   = (const float*)d_in[3];   // [256]
  const float* mod_w  = (const float*)d_in[4];   // [512,512]
  const float* mod_b  = (const float*)d_in[5];   // [512]
  float* out = (float*)d_out;                    // [16,256,64,64]

  // workspace layout (all offsets 256B-aligned), total ~87.9 MB
  char* ws = (char*)d_ws;
  float*          style = (float*)(ws + 0);                  //   32 KB
  float*          Wsq   = (float*)(ws + 32768);              //  512 KB
  float*          coef  = (float*)(ws + 557056);             //   16 KB
  unsigned short* Wt    = (unsigned short*)(ws + 573440);    // 2.25 MB  [9][256][512] bf16
  unsigned short* xs    = (unsigned short*)(ws + 2932736);   // 17.0 MB  [16][33][33][512] bf16
  float*          ypre  = (float*)(ws + 20774912);           // 64.0 MB  [16][256][64][64] f32

  style_kernel <<<32, 256, 0, stream>>>(s, mod_w, mod_b, style);
  prep_w_kernel<<<512, 256, 0, stream>>>(weight, Wt, Wsq);
  xs_kernel    <<<dim3(33, 33, 16), 256, 0, stream>>>(x, style, xs);
  coef_kernel  <<<16, 256, 0, stream>>>(style, Wsq, coef);
  gemm_kernel  <<<dim3(256, 1, 4), 256, 0, stream>>>(xs, Wt, coef, bias, ypre);
  blur_kernel  <<<dim3(4, 4, 4096), dim3(16, 16, 1), 0, stream>>>(ypre, out);
}
